// MambaBlock_67525475827717
// MI455X (gfx1250) — compile-verified, matrix operand order checked by reading, not run on hardware
//
#include <hip/hip_runtime.h>

// ---------------------------------------------------------------------------
// Problem constants (B, N, D) = (2, 4096, 1024)
// ---------------------------------------------------------------------------
constexpr int Bc     = 2;
constexpr int Nseq   = 4096;
constexpr int Dm     = 1024;
constexpr int Din    = 2048;        // D_INNER
constexpr int Dstate = 16;
constexpr int DtRank = 64;
constexpr int Xcols  = DtRank + 2 * Dstate;   // 96
constexpr int Tok    = Bc * Nseq;             // 8192 tokens

typedef _Float16 f16;
typedef __attribute__((ext_vector_type(16))) _Float16 v16h;
typedef __attribute__((ext_vector_type(8)))  _Float16 v8h;
typedef __attribute__((ext_vector_type(8)))  float    v8f;

union V16 { v16h v; v8h h[2]; };

// ---------------------------------------------------------------------------
// CDNA5 async global->LDS copy (16B per lane), with portable fallback.
// Builtin signature (from hipcc diagnostic): pointers to 16-byte int vectors.
// ---------------------------------------------------------------------------
typedef int v4i __attribute__((vector_size(4 * sizeof(int))));
typedef __attribute__((address_space(1))) v4i* gl_v4i_ptr;
typedef __attribute__((address_space(3))) v4i* lds_v4i_ptr;

__device__ __forceinline__ void async_copy16(const f16* g, f16* l) {
#if __has_builtin(__builtin_amdgcn_global_load_async_to_lds_b128)
  __builtin_amdgcn_global_load_async_to_lds_b128(
      (gl_v4i_ptr)(const void*)g,
      (lds_v4i_ptr)(void*)l, 0, 0);
#else
  *reinterpret_cast<uint4*>(l) = *reinterpret_cast<const uint4*>(g);
#endif
}

__device__ __forceinline__ void wait_async() {
#if __has_builtin(__builtin_amdgcn_s_wait_asynccnt)
  __builtin_amdgcn_s_wait_asynccnt(0);
#endif
}

// ---------------------------------------------------------------------------
// f32 -> f16 convert
// ---------------------------------------------------------------------------
__global__ void cvt_f32_f16_kernel(const float* __restrict__ in,
                                   f16* __restrict__ out, int n) {
  int i = blockIdx.x * blockDim.x + threadIdx.x;
  if (i < n) out[i] = (f16)in[i];
}

// ---------------------------------------------------------------------------
// mod = silu(c) @ ada_w.T + ada_b      (B x 3D, tiny)
// ---------------------------------------------------------------------------
__global__ void mod_kernel(const float* __restrict__ c,
                           const float* __restrict__ ada_w,
                           const float* __restrict__ ada_b,
                           float* __restrict__ mod) {
  int j = blockIdx.x * blockDim.x + threadIdx.x;
  if (j >= Bc * 3 * Dm) return;
  int b   = j / (3 * Dm);
  int col = j - b * 3 * Dm;
  const float* cr = c + b * Dm;
  const float* wr = ada_w + (size_t)col * Dm;
  float acc = ada_b[col];
  for (int k = 0; k < Dm; ++k) {
    float cv = cr[k];
    acc += (cv / (1.0f + __expf(-cv))) * wr[k];
  }
  mod[j] = acc;
}

// ---------------------------------------------------------------------------
// LayerNorm + AdaLN modulation -> h (f16). One block per token row.
// ---------------------------------------------------------------------------
__global__ __launch_bounds__(256)
void ln_mod_kernel(const float* __restrict__ x,
                   const float* __restrict__ ln_g,
                   const float* __restrict__ ln_b,
                   const float* __restrict__ mod,
                   f16* __restrict__ h16) {
  const int row = blockIdx.x;            // b*N + n
  const int b   = row / Nseq;
  const float* xr = x + (size_t)row * Dm;
  __shared__ float red[256];

  float s = 0.f;
  for (int i = threadIdx.x; i < Dm; i += 256) s += xr[i];
  red[threadIdx.x] = s; __syncthreads();
  for (int o = 128; o > 0; o >>= 1) {
    if (threadIdx.x < o) red[threadIdx.x] += red[threadIdx.x + o];
    __syncthreads();
  }
  float mu = red[0] * (1.0f / Dm);
  __syncthreads();

  float v = 0.f;
  for (int i = threadIdx.x; i < Dm; i += 256) {
    float d = xr[i] - mu; v += d * d;
  }
  red[threadIdx.x] = v; __syncthreads();
  for (int o = 128; o > 0; o >>= 1) {
    if (threadIdx.x < o) red[threadIdx.x] += red[threadIdx.x + o];
    __syncthreads();
  }
  float rstd = rsqrtf(red[0] * (1.0f / Dm) + 1e-6f);

  const float* shiftp = mod + (size_t)b * 3 * Dm;
  const float* scalep = shiftp + Dm;
  f16* hr = h16 + (size_t)row * Dm;
  for (int i = threadIdx.x; i < Dm; i += 256) {
    float hn = (xr[i] - mu) * rstd * ln_g[i] + ln_b[i];
    hn = hn * (1.0f + scalep[i]) + shiftp[i];
    hr[i] = (f16)hn;
  }
}

// ---------------------------------------------------------------------------
// WMMA GEMM: C[M,N] = A[M,K] @ W[N,K]^T   (A,W f16 row-major, C f32)
// Block tile 128x128, BK=32, 8 waves (4 over M x 2 over N), each wave 32x64
// (2x4 WMMA tiles -> 8 v_wmma per K-step, 12 ds_load_b128).
// Double-buffered LDS fed by GLOBAL_LOAD_ASYNC_TO_LDS_B128 (ASYNCcnt),
// prefetch of tile kt+2.
// ---------------------------------------------------------------------------
#define BM 128
#define BN 128
#define BK 32
#define LDP (BK + 8)

__global__ __launch_bounds__(256)
void gemm_f16_kernel(const f16* __restrict__ A, int lda,
                     const f16* __restrict__ W, int ldw,
                     float* __restrict__ C, int ldc,
                     int M, int N, int K) {
  __shared__ f16 As[2][BM][LDP];
  __shared__ f16 Ws[2][BN][LDP];

  const int t    = threadIdx.x;
  const int lane = t & 31;
  const int wave = t >> 5;
  const int wy   = wave & 3;    // M sub-tile (4 x 32 rows)
  const int wx   = wave >> 2;   // N sub-tile (2 x 64 cols)
  const int mBase = blockIdx.y * BM;
  const int nBase = blockIdx.x * BN;

  auto loadTile = [&](int kt, int stage) {
    const int k0 = kt * BK;
    // A: 128x32 f16 = 512 chunks of 8 f16 (16B); 2 per thread
#pragma unroll
    for (int rep = 0; rep < 2; ++rep) {
      int cid = t + rep * 256;
      int row = cid >> 2;
      int cc  = (cid & 3) * 8;
      f16* dst = &As[stage][row][cc];
      int gr = mBase + row;
      if (gr < M) async_copy16(A + (size_t)gr * lda + k0 + cc, dst);
      else        *reinterpret_cast<uint4*>(dst) = make_uint4(0u, 0u, 0u, 0u);
    }
    // W: 128x32 f16 = 512 chunks; 2 per thread
#pragma unroll
    for (int rep = 0; rep < 2; ++rep) {
      int cid = t + rep * 256;
      int row = cid >> 2;
      int cc  = (cid & 3) * 8;
      f16* dst = &Ws[stage][row][cc];
      int gr = nBase + row;
      if (gr < N) async_copy16(W + (size_t)gr * ldw + k0 + cc, dst);
      else        *reinterpret_cast<uint4*>(dst) = make_uint4(0u, 0u, 0u, 0u);
    }
  };

  // Fragment builders: ISA 16-bit 16x32 layout. Lanes 0-15: K 0-7 & 16-23,
  // lanes 16-31: K 8-15 & 24-31, row/col = lane&15.
  auto fragA = [&](int stage, int mo) -> v16h {
    int row = wy * 32 + mo + (lane & 15);
    int kh  = (lane >> 4) * 8;
    V16 r;
    r.h[0] = *reinterpret_cast<const v8h*>(&As[stage][row][kh]);
    r.h[1] = *reinterpret_cast<const v8h*>(&As[stage][row][kh + 16]);
    return r.v;
  };
  auto fragW = [&](int stage, int no) -> v16h {
    int row = wx * 64 + no + (lane & 15);
    int kh  = (lane >> 4) * 8;
    V16 r;
    r.h[0] = *reinterpret_cast<const v8h*>(&Ws[stage][row][kh]);
    r.h[1] = *reinterpret_cast<const v8h*>(&Ws[stage][row][kh + 16]);
    return r.v;
  };

  v8f acc[2][4];
#pragma unroll
  for (int i = 0; i < 2; ++i)
#pragma unroll
    for (int j = 0; j < 4; ++j) acc[i][j] = (v8f){};

  const int ntiles = K / BK;

  loadTile(0, 0);
  wait_async();
  __syncthreads();

  for (int kt = 0; kt < ntiles; ++kt) {
    const int cur = kt & 1;
    if (kt + 1 < ntiles) loadTile(kt + 1, cur ^ 1);
    if (kt + 2 < ntiles) {
      int ra = mBase + (t >> 1); if (ra >= M) ra = M - 1;
      int rw = nBase + (t >> 1); if (rw >= N) rw = N - 1;
      __builtin_prefetch(A + (size_t)ra * lda + (kt + 2) * BK, 0, 1);
      __builtin_prefetch(W + (size_t)rw * ldw + (kt + 2) * BK, 0, 1);
    }
    v16h a[2], b[4];
#pragma unroll
    for (int i = 0; i < 2; ++i) a[i] = fragA(cur, i * 16);
#pragma unroll
    for (int j = 0; j < 4; ++j) b[j] = fragW(cur, j * 16);
#pragma unroll
    for (int i = 0; i < 2; ++i)
#pragma unroll
      for (int j = 0; j < 4; ++j)
        acc[i][j] = __builtin_amdgcn_wmma_f32_16x16x32_f16(
            false, a[i], false, b[j], (short)0, acc[i][j], false, false);
    wait_async();
    __syncthreads();
  }

  // C layout: VGPR r -> M = r (lanes 0-15) / r+8 (lanes 16-31); N = lane&15.
#pragma unroll
  for (int i = 0; i < 2; ++i) {
#pragma unroll
    for (int j = 0; j < 4; ++j) {
      int col = nBase + wx * 64 + j * 16 + (lane & 15);
      if (col >= N) continue;
      int rbase = mBase + wy * 32 + i * 16 + ((lane >> 4) << 3);
#pragma unroll
      for (int r = 0; r < 8; ++r) {
        int row = rbase + r;
        if (row < M) C[(size_t)row * ldc + col] = acc[i][j][r];
      }
    }
  }
}

// ---------------------------------------------------------------------------
// Depthwise causal conv (width 4) + bias + SiLU  -> u (f16)
// u_pre comes from columns [0, Din) of xz (f32).
// ---------------------------------------------------------------------------
__global__ void conv_silu_kernel(const float* __restrict__ xz,
                                 const float* __restrict__ conv_w,
                                 const float* __restrict__ conv_b,
                                 f16* __restrict__ uconv) {
  size_t i = (size_t)blockIdx.x * blockDim.x + threadIdx.x;
  if (i >= (size_t)Tok * Din) return;
  int d   = (int)(i % Din);
  int n   = (int)((i / Din) % Nseq);
  int b   = (int)(i / ((size_t)Nseq * Din));
  float acc = conv_b[d];
#pragma unroll
  for (int j = 0; j < 4; ++j) {
    int tpos = n - 3 + j;
    if (tpos >= 0)
      acc += xz[((size_t)(b * Nseq + tpos)) * (2 * Din) + d] * conv_w[d * 4 + j];
  }
  acc = acc / (1.0f + __expf(-acc));
  uconv[i] = (f16)acc;
}

// ---------------------------------------------------------------------------
// dt softplus (in-place):  dt = softplus(dt_raw + dt_b[d])
// ---------------------------------------------------------------------------
__global__ void dt_softplus_kernel(float* __restrict__ dt,
                                   const float* __restrict__ dt_b) {
  size_t i = (size_t)blockIdx.x * blockDim.x + threadIdx.x;
  if (i >= (size_t)Tok * Din) return;
  int d = (int)(i % Din);
  float xv = dt[i] + dt_b[d];
  dt[i] = (xv > 20.0f) ? xv : log1pf(__expf(xv));
}

// ---------------------------------------------------------------------------
// Selective scan: one lane per (b, d) channel; 16 states in registers;
// per-timestep B/C broadcast via LDS. Fuses +u*D_skip and *silu(z).
// ---------------------------------------------------------------------------
__global__ __launch_bounds__(256)
void scan_kernel(const float* __restrict__ dbc,
                 const float* __restrict__ dt,
                 const f16*   __restrict__ uconv,
                 const float* __restrict__ xz,
                 const float* __restrict__ A_log,
                 const float* __restrict__ D_skip,
                 f16* __restrict__ y16) {
  const int b = blockIdx.x >> 3;                       // 8 blocks per batch
  const int d = ((blockIdx.x & 7) << 8) + threadIdx.x; // channel
  float A[Dstate], h[Dstate];
#pragma unroll
  for (int s = 0; s < Dstate; ++s) {
    A[s] = -__expf(A_log[d * Dstate + s]);
    h[s] = 0.0f;
  }
  const float Dsk = D_skip[d];
  __shared__ float sBC[2 * Dstate];
  const size_t base = (size_t)b * Nseq;

  for (int n = 0; n < Nseq; ++n) {
    if (threadIdx.x < 2 * Dstate)
      sBC[threadIdx.x] = dbc[(base + n) * Xcols + DtRank + threadIdx.x];
    __syncthreads();

    const size_t idx = (base + n) * (size_t)Din + d;
    const float dtv = dt[idx];
    const float uv  = (float)uconv[idx];
    const float du  = dtv * uv;
    const float zv  = xz[(base + n) * (size_t)(2 * Din) + Din + d];
    float y = 0.0f;
#pragma unroll
    for (int s = 0; s < Dstate; ++s) {
      float dA = __expf(dtv * A[s]);
      h[s] = dA * h[s] + du * sBC[s];
      y += h[s] * sBC[Dstate + s];
    }
    y += uv * Dsk;
    y *= zv / (1.0f + __expf(-zv));
    y16[idx] = (f16)y;
    __syncthreads();
  }
}

// ---------------------------------------------------------------------------
// final: out = x + gate * tmp
// ---------------------------------------------------------------------------
__global__ void final_kernel(const float* __restrict__ x,
                             const float* __restrict__ mod,
                             const float* __restrict__ tmp,
                             float* __restrict__ out) {
  size_t i = (size_t)blockIdx.x * blockDim.x + threadIdx.x;
  if (i >= (size_t)Tok * Dm) return;
  int col = (int)(i % Dm);
  int b   = (int)(i / ((size_t)Nseq * Dm));
  float g = mod[(size_t)b * 3 * Dm + 2 * Dm + col];
  out[i] = x[i] + g * tmp[i];
}

// ---------------------------------------------------------------------------
// Host-side launch
// ---------------------------------------------------------------------------
static inline size_t align_up(size_t v) { return (v + 255) & ~size_t(255); }

extern "C" void kernel_launch(void* const* d_in, const int* in_sizes, int n_in,
                              void* d_out, int out_size, void* d_ws, size_t ws_size,
                              hipStream_t stream) {
  (void)in_sizes; (void)n_in; (void)out_size; (void)ws_size;

  const float* x       = (const float*)d_in[0];
  const float* c       = (const float*)d_in[1];
  const float* ln_g    = (const float*)d_in[2];
  const float* ln_b    = (const float*)d_in[3];
  const float* ada_w   = (const float*)d_in[4];
  const float* ada_b   = (const float*)d_in[5];
  const float* in_w    = (const float*)d_in[6];
  const float* conv_w  = (const float*)d_in[7];
  const float* conv_b  = (const float*)d_in[8];
  const float* xproj_w = (const float*)d_in[9];
  const float* dt_w    = (const float*)d_in[10];
  const float* dt_b    = (const float*)d_in[11];
  const float* A_log   = (const float*)d_in[12];
  const float* D_skip  = (const float*)d_in[13];
  const float* out_w   = (const float*)d_in[14];
  float* outp          = (float*)d_out;

  // Workspace layout
  char* ws = (char*)d_ws;
  size_t off = 0;
  auto carve = [&](size_t bytes) { char* p = ws + off; off += align_up(bytes); return (void*)p; };

  float* mod     = (float*)carve((size_t)Bc * 3 * Dm * 4);
  f16*   h16     = (f16*)  carve((size_t)Tok * Dm * 2);
  f16*   w_in16  = (f16*)  carve((size_t)2 * Din * Dm * 2);
  f16*   w_xp16  = (f16*)  carve((size_t)Xcols * Din * 2);
  f16*   w_dt16  = (f16*)  carve((size_t)Din * DtRank * 2);
  f16*   w_out16 = (f16*)  carve((size_t)Dm * Din * 2);
  float* xz      = (float*)carve((size_t)Tok * 2 * Din * 4);   // also reused as out_tmp
  f16*   uconv16 = (f16*)  carve((size_t)Tok * Din * 2);
  float* dbc     = (float*)carve((size_t)Tok * Xcols * 4);
  f16*   dbc16   = (f16*)  carve((size_t)Tok * Xcols * 2);
  float* dtbuf   = (float*)carve((size_t)Tok * Din * 4);
  f16*   y16     = (f16*)  carve((size_t)Tok * Din * 2);

  const int TB = 256;
  auto blocks = [&](size_t n) { return (unsigned)((n + TB - 1) / TB); };

  // 1) weight conversions to f16
  cvt_f32_f16_kernel<<<blocks((size_t)2 * Din * Dm), TB, 0, stream>>>(in_w,    w_in16,  2 * Din * Dm);
  cvt_f32_f16_kernel<<<blocks((size_t)Xcols * Din),  TB, 0, stream>>>(xproj_w, w_xp16,  Xcols * Din);
  cvt_f32_f16_kernel<<<blocks((size_t)Din * DtRank), TB, 0, stream>>>(dt_w,    w_dt16,  Din * DtRank);
  cvt_f32_f16_kernel<<<blocks((size_t)Dm * Din),     TB, 0, stream>>>(out_w,   w_out16, Dm * Din);

  // 2) AdaLN modulation vector
  mod_kernel<<<blocks((size_t)Bc * 3 * Dm), TB, 0, stream>>>(c, ada_w, ada_b, mod);

  // 3) LayerNorm + modulation -> h16
  ln_mod_kernel<<<Tok, TB, 0, stream>>>(x, ln_g, ln_b, mod, h16);

  // 4) xz = h @ in_w^T    (M=8192, N=4096, K=1024)
  {
    dim3 grid((2 * Din) / BN, Tok / BM);
    gemm_f16_kernel<<<grid, TB, 0, stream>>>(h16, Dm, w_in16, Dm, xz, 2 * Din,
                                             Tok, 2 * Din, Dm);
  }

  // 5) depthwise conv + SiLU -> uconv16
  conv_silu_kernel<<<blocks((size_t)Tok * Din), TB, 0, stream>>>(xz, conv_w, conv_b, uconv16);

  // 6) dbc = u @ xproj_w^T  (M=8192, N=96, K=2048)
  {
    dim3 grid((Xcols + BN - 1) / BN, Tok / BM);
    gemm_f16_kernel<<<grid, TB, 0, stream>>>(uconv16, Din, w_xp16, Din, dbc, Xcols,
                                             Tok, Xcols, Din);
  }

  // 7) dbc f32 -> f16 (GEMM input for dt projection)
  cvt_f32_f16_kernel<<<blocks((size_t)Tok * Xcols), TB, 0, stream>>>(dbc, dbc16, Tok * Xcols);

  // 8) dt_raw = dbc[:, :64] @ dt_w^T  (M=8192, N=2048, K=64; lda=96)
  {
    dim3 grid(Din / BN, Tok / BM);
    gemm_f16_kernel<<<grid, TB, 0, stream>>>(dbc16, Xcols, w_dt16, DtRank, dtbuf, Din,
                                             Tok, Din, DtRank);
  }

  // 9) dt = softplus(dt_raw + dt_b)
  dt_softplus_kernel<<<blocks((size_t)Tok * Din), TB, 0, stream>>>(dtbuf, dt_b);

  // 10) selective scan -> y16
  scan_kernel<<<Bc * (Din / TB), TB, 0, stream>>>(dbc, dtbuf, uconv16, xz, A_log, D_skip, y16);

  // 11) tmp = y @ out_w^T  (M=8192, N=1024, K=2048) — reuse xz region as tmp
  float* tmp = xz;
  {
    dim3 grid(Dm / BN, Tok / BM);
    gemm_f16_kernel<<<grid, TB, 0, stream>>>(y16, Din, w_out16, Din, tmp, Dm,
                                             Tok, Dm, Din);
  }

  // 12) out = x + gate * tmp
  final_kernel<<<blocks((size_t)Tok * Dm), TB, 0, stream>>>(x, mod, tmp, outp);
}
